// LukeSelfAttention_49598282334450
// MI455X (gfx1250) — compile-verified, matrix operand order checked by reading, not run on hardware
//
#include <hip/hip_runtime.h>

// ---------------------------------------------------------------------------
// LUKE self-attention for gfx1250 (wave32, WMMA f32_16x16x32_f16,
// async global->LDS weight staging, flash-style fused attention)
// ---------------------------------------------------------------------------
#define B_      8
#define SW_     512
#define SE_     128
#define S_      640          // SW_ + SE_
#define H_      1024
#define HEADS_  16
#define D_      64
#define SCALE_  0.125f       // 1/sqrt(64)

#define WLD_    72           // padded LDS row stride (halves) for weight tile
#define PLD_    40           // padded LDS row stride (halves) for P tile

typedef _Float16 half_t;
typedef __attribute__((ext_vector_type(16))) _Float16 v16h;
typedef __attribute__((ext_vector_type(8)))  _Float16 v8h;
typedef __attribute__((ext_vector_type(8)))  float    v8f;

// --- fragment loaders (ISA 7.12.2 layouts) ---------------------------------
// A-fragment (16x32 f16, row-major source, stride ld):
//   lane L (<16): row M=L, K = {0..7, 16..23}; lane L+16: row M=L, K = {8..15, 24..31}
// B-fragment for Bt source (B[k][n] = M[n*ld + k]) uses the mirrored pattern.
__device__ __forceinline__ v16h load_frag16(const half_t* __restrict__ p, int ld, int lane) {
  const int rc = lane & 15;            // M for A, N for B
  const int kb = (lane >> 4) << 3;     // 0 or 8
  const half_t* base = p + (size_t)rc * ld + kb;
  v8h lo = *(const v8h*)(base);        // K = kb + 0..7
  v8h hi = *(const v8h*)(base + 16);   // K = kb + 16..23
  return __builtin_shufflevector(lo, hi, 0,1,2,3,4,5,6,7,8,9,10,11,12,13,14,15);
}

__device__ __forceinline__ v8f wmma32(v16h a, v16h b, v8f c) {
  return __builtin_amdgcn_wmma_f32_16x16x32_f16(false, a, false, b, (short)0, c, false, false);
}

// async global -> LDS, 16B per lane, ASYNCcnt-tracked (CDNA5)
__device__ __forceinline__ void async_load_b128(unsigned lds_byte_addr,
                                                unsigned global_voffset,
                                                const half_t* __restrict__ sbase) {
  asm volatile("global_load_async_to_lds_b128 %0, %1, %2"
               :: "v"(lds_byte_addr), "v"(global_voffset), "s"(sbase)
               : "memory");
}
#define ASYNC_WAIT(n) asm volatile("s_wait_asynccnt " #n ::: "memory")

// --- pack fp32 -> fp16 ------------------------------------------------------
__global__ void pack_x_kernel(const float* __restrict__ w, const float* __restrict__ e,
                              half_t* __restrict__ x16) {
  size_t idx = (size_t)blockIdx.x * blockDim.x + threadIdx.x;
  if (idx >= (size_t)B_ * S_ * H_) return;
  int b = (int)(idx / ((size_t)S_ * H_));
  int r = (int)(idx % ((size_t)S_ * H_));
  int s = r / H_, i = r % H_;
  float v = (s < SW_) ? w[((size_t)b * SW_ + s) * H_ + i]
                      : e[((size_t)b * SE_ + (s - SW_)) * H_ + i];
  x16[idx] = (half_t)v;
}

__global__ void pack_w_kernel(const float* __restrict__ w0, const float* __restrict__ w1,
                              const float* __restrict__ w2, const float* __restrict__ w3,
                              const float* __restrict__ w4, const float* __restrict__ w5,
                              half_t* __restrict__ dst) {
  size_t idx = (size_t)blockIdx.x * blockDim.x + threadIdx.x;
  const size_t N = (size_t)H_ * H_;
  if (idx >= 6 * N) return;
  int which = (int)(idx / N);
  size_t off = idx % N;
  const float* src = which == 0 ? w0 : which == 1 ? w1 : which == 2 ? w2
                   : which == 3 ? w3 : which == 4 ? w4 : w5;
  dst[idx] = (half_t)src[off];
}

// --- projection GEMM: out[b][s][o] = sum_i X[b][s][i] * W[o][i] + bias[o] ---
// Block = 4 waves, 128x64 output tile; wave = 32x64 (16 WMMA per 64-k-step,
// every LDS B-fragment feeds 2 WMMAs). Weight tile (8KB) staged through a
// double-buffered, bank-conflict-padded LDS tile via
// global_load_async_to_lds_b128; A fragments ping-pong between two register
// sets (no rotation moves). k-loop steps 128 = two explicit pipeline stages.
// out_mode 0: row-major half out[b][s][H]; out_mode 1: transposed Vt[b][o][S].
__global__ void __launch_bounds__(128)
proj_gemm_kernel(const half_t* __restrict__ X,
                 const half_t* __restrict__ Wp, const half_t* __restrict__ Ws,
                 const float* __restrict__ bp, const float* __restrict__ bs,
                 half_t* __restrict__ out, int out_mode) {
  __shared__ half_t sW[2][64 * WLD_];   // padded: bank = 36*lane % 64, conflict-free

  const int b    = blockIdx.z;
  const int tid  = threadIdx.x;
  const int wave = tid >> 5;
  const int lane = tid & 31;
  const int rblk = blockIdx.y * 128;               // 128 | 512 -> per-block side
  const int row0 = rblk + wave * 32;
  const int col0 = blockIdx.x * 64;

  const half_t* Xb   = X + ((size_t)b * S_ + row0) * H_;
  const half_t* W    = (rblk < SW_) ? Wp : Ws;     // uniform per block
  const float*  bias = (rblk < SW_) ? bp : bs;

  const unsigned ldsBase = (unsigned)(size_t)(void*)&sW[0][0];

  // one 8KB weight stage: rows col0..col0+63, halves k..k+63 (4 x 2KB issues)
  auto issue_stage = [&](int k, int buf) {
#pragma unroll
    for (int q = 0; q < 4; ++q) {
      const int c   = q * 128 + tid;     // 0..511 16B-chunks
      const int row = c >> 3;            // 0..63  (output column o - col0)
      const int c16 = c & 7;             // 16B chunk within 128B of row data
      const unsigned voff = (unsigned)((((col0 + row) * H_ + k) << 1) + (c16 << 4));
      const unsigned ldsa = ldsBase + (unsigned)(buf * (64 * WLD_ * 2) + row * (WLD_ * 2) + c16 * 16);
      async_load_b128(ldsa, voff, W);
    }
  };

  // A fragments: [rowGroup(2)][kHalf(2)]
  auto load_a = [&](int k, v16h* dst) {
    dst[0] = load_frag16(Xb + k,               H_, lane);
    dst[1] = load_frag16(Xb + k + 32,          H_, lane);
    dst[2] = load_frag16(Xb + 16 * H_ + k,      H_, lane);
    dst[3] = load_frag16(Xb + 16 * H_ + k + 32, H_, lane);
  };

  v8f acc[2][4] = {};
  auto compute = [&](const v16h* a, int buf) {
#pragma unroll
    for (int j = 0; j < 4; ++j) {
      v16h b0 = load_frag16(&sW[buf][j * 16 * WLD_],      WLD_, lane);
      v16h b1 = load_frag16(&sW[buf][j * 16 * WLD_ + 32], WLD_, lane);
      acc[0][j] = wmma32(a[0], b0, acc[0][j]);
      acc[0][j] = wmma32(a[1], b1, acc[0][j]);
      acc[1][j] = wmma32(a[2], b0, acc[1][j]);
      acc[1][j] = wmma32(a[3], b1, acc[1][j]);
    }
  };

  v16h aA[4], aB[4];
  issue_stage(0, 0);
  load_a(0, aA);

  for (int k = 0; k < H_; k += 128) {
    // stage A: consume buf0 (k), fill buf1 (k+64)  [k+64 < 1024 always]
    issue_stage(k + 64, 1);
    load_a(k + 64, aB);
    ASYNC_WAIT(0x4);
    __syncthreads();
    compute(aA, 0);
    __syncthreads();

    // stage B: consume buf1 (k+64), fill buf0 (k+128)
    const bool more = (k + 128 < H_);
    if (more) { issue_stage(k + 128, 0); load_a(k + 128, aA); }
    if (more) { ASYNC_WAIT(0x4); } else { ASYNC_WAIT(0x0); }
    __syncthreads();
    compute(aB, 1);
    __syncthreads();
  }

  const int mloc = (lane >> 4) << 3;   // 0 or 8
  const int n    = lane & 15;
#pragma unroll
  for (int i = 0; i < 2; ++i) {
    const int rbase = row0 + i * 16 + mloc;
    if (out_mode == 0) {
#pragma unroll
      for (int j = 0; j < 4; ++j) {
        const int o = col0 + j * 16 + n;
        const float bv = bias[o];
        half_t* op = out + ((size_t)b * S_ + rbase) * H_ + o;
#pragma unroll
        for (int r = 0; r < 8; ++r) op[(size_t)r * H_] = (half_t)(acc[i][j][r] + bv);
      }
    } else {
#pragma unroll
      for (int j = 0; j < 4; ++j) {
        const int o = col0 + j * 16 + n;
        const float bv = bias[o];
        v8h pk;
#pragma unroll
        for (int r = 0; r < 8; ++r) pk[r] = (half_t)(acc[i][j][r] + bv);
        *(v8h*)(out + ((size_t)b * H_ + o) * S_ + rbase) = pk;  // 16B store
      }
    }
  }
}

// --- fused attention: scores -> online softmax -> P.V ----------------------
// One wave per (b, head, 16-query tile). Qw = query-vs-word-keys tensor,
// Qe = query-vs-entity-keys tensor (already per-row weight-selected).
__global__ void __launch_bounds__(128)
attn_kernel(const half_t* __restrict__ Qw, const half_t* __restrict__ Qe,
            const half_t* __restrict__ K,  const half_t* __restrict__ Vt,
            const float* __restrict__ mask, float* __restrict__ out) {
  __shared__ half_t sP[4 * 16 * PLD_];   // per-wave 16xPLD_ (bank-padded)

  const int b    = blockIdx.z;
  const int h    = blockIdx.y;
  const int wave = threadIdx.x >> 5;
  const int lane = threadIdx.x & 31;
  const int q0   = (blockIdx.x * 4 + wave) * 16;

  const half_t* Qwp = Qw + ((size_t)b * S_ + q0) * H_ + h * D_;
  const half_t* Qep = Qe + ((size_t)b * S_ + q0) * H_ + h * D_;
  const half_t* Kb  = K  + (size_t)b * S_ * H_ + h * D_;
  const half_t* Vb  = Vt + ((size_t)b * H_ + h * D_) * S_;
  const float*  mb  = mask + (size_t)b * S_;
  half_t* myP = &sP[wave * 16 * PLD_];

  // query fragments (d 0..31 / 32..63), both key-side variants
  v16h aW0 = load_frag16(Qwp,      H_, lane);
  v16h aW1 = load_frag16(Qwp + 32, H_, lane);
  v16h aE0 = load_frag16(Qep,      H_, lane);
  v16h aE1 = load_frag16(Qep + 32, H_, lane);

  float run_max[8], run_sum[8];
#pragma unroll
  for (int r = 0; r < 8; ++r) { run_max[r] = -1e30f; run_sum[r] = 0.0f; }
  v8f ctx[4] = {v8f{}, v8f{}, v8f{}, v8f{}};

  const int prow = (lane >> 4) << 3;
  const int n    = lane & 15;

  for (int t = 0; t < S_; t += 32) {
    if (t + 32 < S_) __builtin_prefetch(Kb + (size_t)(t + 32) * H_, 0, 0);

    const bool wordKeys = (t < SW_);            // 512 % 32 == 0 -> no mixing
    v16h a0 = wordKeys ? aW0 : aE0;
    v16h a1 = wordKeys ? aW1 : aE1;

    // two 16x16 score tiles (keys t..t+15, t+16..t+31), K-dim = d = 64
    v8f s0 = v8f{}, s1 = v8f{};
    {
      v16h bk0 = load_frag16(Kb + (size_t)t * H_,        H_, lane);
      v16h bk1 = load_frag16(Kb + (size_t)t * H_ + 32,   H_, lane);
      s0 = wmma32(a0, bk0, s0);  s0 = wmma32(a1, bk1, s0);
      v16h bk2 = load_frag16(Kb + (size_t)(t + 16) * H_,      H_, lane);
      v16h bk3 = load_frag16(Kb + (size_t)(t + 16) * H_ + 32, H_, lane);
      s1 = wmma32(a0, bk2, s1);  s1 = wmma32(a1, bk3, s1);
    }

    const float m0 = mb[t + n];
    const float m1 = mb[t + 16 + n];

    // online softmax per row (rows live in lane-halves per C layout)
#pragma unroll
    for (int r = 0; r < 8; ++r) {
      float x0 = s0[r] * SCALE_ + m0;
      float x1 = s1[r] * SCALE_ + m1;
      float mx = fmaxf(x0, x1);
#pragma unroll
      for (int o = 1; o < 16; o <<= 1) mx = fmaxf(mx, __shfl_xor(mx, o));
      float nm    = fmaxf(run_max[r], mx);
      float alpha = __expf(run_max[r] - nm);
      run_max[r]  = nm;
      float p0 = __expf(x0 - nm);
      float p1 = __expf(x1 - nm);
      float rs = p0 + p1;
#pragma unroll
      for (int o = 1; o < 16; o <<= 1) rs += __shfl_xor(rs, o);
      run_sum[r] = run_sum[r] * alpha + rs;
#pragma unroll
      for (int j = 0; j < 4; ++j) ctx[j][r] *= alpha;
      s0[r] = p0; s1[r] = p1;
    }

    // C-layout -> A-layout via per-wave LDS tile (same-wave, no barrier)
#pragma unroll
    for (int r = 0; r < 8; ++r) {
      myP[(prow + r) * PLD_ + n]      = (half_t)s0[r];
      myP[(prow + r) * PLD_ + 16 + n] = (half_t)s1[r];
    }
    v16h aP = load_frag16(myP, PLD_, lane);   // P as 16x32 A fragment

    // ctx += P (16x32) x V (32x16) for the 4 d-column tiles
#pragma unroll
    for (int j = 0; j < 4; ++j) {
      v16h bv = load_frag16(Vb + (size_t)(j * 16) * S_ + t, S_, lane);
      ctx[j] = wmma32(aP, bv, ctx[j]);
    }
  }

  // normalize and scatter to (word ctx ++ entity ctx) fp32 output
  const size_t entBase = (size_t)B_ * SW_ * H_;
#pragma unroll
  for (int j = 0; j < 4; ++j) {
#pragma unroll
    for (int r = 0; r < 8; ++r) {
      const int s = q0 + prow + r;
      const int d = h * D_ + j * 16 + n;
      const float val = ctx[j][r] / run_sum[r];
      size_t off = (s < SW_)
          ? (((size_t)b * SW_ + s) * H_ + d)
          : (entBase + ((size_t)b * SE_ + (s - SW_)) * H_ + d);
      out[off] = val;
    }
  }
}

// ---------------------------------------------------------------------------
extern "C" void kernel_launch(void* const* d_in, const int* in_sizes, int n_in,
                              void* d_out, int out_size, void* d_ws, size_t ws_size,
                              hipStream_t stream) {
  (void)in_sizes; (void)n_in; (void)out_size; (void)ws_size;

  const float* wrd  = (const float*)d_in[0];
  const float* ent  = (const float*)d_in[1];
  const float* msk  = (const float*)d_in[2];
  const float* Wq   = (const float*)d_in[3];  const float* bq   = (const float*)d_in[4];
  const float* Wk   = (const float*)d_in[5];  const float* bk   = (const float*)d_in[6];
  const float* Wv   = (const float*)d_in[7];  const float* bv   = (const float*)d_in[8];
  const float* Ww2e = (const float*)d_in[9];  const float* bw2e = (const float*)d_in[10];
  const float* We2w = (const float*)d_in[11]; const float* be2w = (const float*)d_in[12];
  const float* We2e = (const float*)d_in[13]; const float* be2e = (const float*)d_in[14];

  const size_t NX = (size_t)B_ * S_ * H_;      // 5,242,880 halves per act buffer
  const size_t NW = (size_t)H_ * H_;           // 1,048,576 halves per weight
  half_t* X16   = (half_t*)d_ws;
  half_t* K16   = X16  + NX;
  half_t* Qw16  = K16  + NX;
  half_t* Qe16  = Qw16 + NX;
  half_t* Vt16  = Qe16 + NX;                   // transposed [b][o][S]
  half_t* Wq16  = Vt16 + NX;
  half_t* Wk16  = Wq16 + NW;
  half_t* Wv16  = Wk16 + NW;
  half_t* Ww2e16= Wv16 + NW;
  half_t* We2w16= Ww2e16 + NW;
  half_t* We2e16= We2w16 + NW;

  // stage 0: fp32 -> fp16 packing
  pack_x_kernel<<<(unsigned)(NX / 256), 256, 0, stream>>>(wrd, ent, X16);
  pack_w_kernel<<<(unsigned)(6 * NW / 256), 256, 0, stream>>>(Wq, Wk, Wv, Ww2e, We2w, We2e, Wq16);

  // stage 1: projection GEMMs (async-LDS weight pipeline, 128x64 block tiles)
  dim3 ggrid(H_ / 64, S_ / 128, B_);           // (16,5,8), 128 threads
  proj_gemm_kernel<<<ggrid, 128, 0, stream>>>(X16, Wk16,   Wk16,   bk,   bk,   K16,  0);
  proj_gemm_kernel<<<ggrid, 128, 0, stream>>>(X16, Wq16,   We2w16, bq,   be2w, Qw16, 0);  // vs word keys
  proj_gemm_kernel<<<ggrid, 128, 0, stream>>>(X16, Ww2e16, We2e16, bw2e, be2e, Qe16, 0);  // vs entity keys
  proj_gemm_kernel<<<ggrid, 128, 0, stream>>>(X16, Wv16,   Wv16,   bv,   bv,   Vt16, 1);  // transposed V

  // stage 2: fused attention (flash-style online softmax)
  dim3 agrid(S_ / 64, HEADS_, B_);             // (10,16,8), 4 waves/block
  attn_kernel<<<agrid, 128, 0, stream>>>(Qw16, Qe16, K16, Vt16, msk, (float*)d_out);
}
